// RecurrentFlowUnit_6906307412194
// MI455X (gfx1250) — matrix operand order repaired and checked by reading, not compiled, not run on hardware
//
#include <hip/hip_runtime.h>

typedef __attribute__((ext_vector_type(16))) _Float16 v16h;
typedef __attribute__((ext_vector_type(8)))  _Float16 v8h;
typedef __attribute__((ext_vector_type(8)))  float    v8f;

#define HW_ (48 * 64)

// ---------------------------------------------------------------------------
// Weight permute: f32 OIHW -> f16 [OC][tap*IC + ic], zero-padded rows oc>=OCV.
// src index = (oc*IC + ic)*TAPS + tap ; dst index = oc*(IC*TAPS) + tap*IC + ic
// ---------------------------------------------------------------------------
__global__ void permute_w_f16(const float* __restrict__ src, _Float16* __restrict__ dst,
                              int IC, int TAPS, int OCV, int ndst) {
  const int i = blockIdx.x * blockDim.x + threadIdx.x;
  if (i >= ndst) return;
  const int ICT = IC * TAPS;
  const int oc = i / ICT;
  const int r  = i % ICT;
  const int tap = r / IC;
  const int ic  = r % IC;
  const float v = (oc < OCV) ? src[((size_t)(oc * IC + ic)) * TAPS + tap] : 0.0f;
  dst[i] = (_Float16)v;
}

// ---------------------------------------------------------------------------
// Warp feat2 at coords + (du,dv) for 49 displacements, concat with feat1,
// emit f16 feat[98][128][48*64].  n = (b*7 + v)*7 + u, dv->y, du->x.
// Bilinear, zeros padding, raw pixel coords (align_corners=True).
// ---------------------------------------------------------------------------
__global__ void warp_concat_kernel(const float* __restrict__ feat1,
                                   const float* __restrict__ feat2,
                                   const float* __restrict__ coords,
                                   _Float16* __restrict__ feat) {
  const int idx = blockIdx.x * blockDim.x + threadIdx.x;
  if (idx >= 98 * HW_) return;
  const int p = idx % HW_;
  const int n = idx / HW_;
  const int u = n % 7;
  const int v = (n / 7) % 7;
  const int b = n / 49;

  const float xs = coords[((size_t)b * 2 + 0) * HW_ + p] + (float)(u - 3);
  const float ys = coords[((size_t)b * 2 + 1) * HW_ + p] + (float)(v - 3);
  const float fx0 = floorf(xs), fy0 = floorf(ys);
  const int x0 = (int)fx0, y0 = (int)fy0;
  const float wx = xs - fx0, wy = ys - fy0;
  const int x1 = x0 + 1, y1 = y0 + 1;
  const bool vx0 = (x0 >= 0) && (x0 < 64), vx1 = (x1 >= 0) && (x1 < 64);
  const bool vy0 = (y0 >= 0) && (y0 < 48), vy1 = (y1 >= 0) && (y1 < 48);
  auto cl = [](int a, int lim) { return a < 0 ? 0 : (a > lim ? lim : a); };
  const int xa = cl(x0, 63), xb = cl(x1, 63);
  const int ya = cl(y0, 47), yb = cl(y1, 47);
  const float w00 = (1.f - wx) * (1.f - wy) * ((vx0 && vy0) ? 1.f : 0.f);
  const float w01 = wx * (1.f - wy) * ((vx1 && vy0) ? 1.f : 0.f);
  const float w10 = (1.f - wx) * wy * ((vx0 && vy1) ? 1.f : 0.f);
  const float w11 = wx * wy * ((vx1 && vy1) ? 1.f : 0.f);
  const int i00 = ya * 64 + xa, i01 = ya * 64 + xb;
  const int i10 = yb * 64 + xa, i11 = yb * 64 + xb;

  const size_t f1b = ((size_t)b * 64) * HW_;
  const size_t ob  = (size_t)n * 128 * HW_ + p;
  for (int c = 0; c < 64; ++c) {
    feat[ob + (size_t)c * HW_] = (_Float16)feat1[f1b + (size_t)c * HW_ + p];
    const float* f2c = feat2 + f1b + (size_t)c * HW_;
    const float acc = w00 * f2c[i00] + w01 * f2c[i01] + w10 * f2c[i10] + w11 * f2c[i11];
    feat[ob + (size_t)(64 + c) * HW_] = (_Float16)acc;
  }
}

// ---------------------------------------------------------------------------
// Fragment builders (CDNA5 ISA VGPR layouts, CDNA5 ISA §7.12.2).
// A 16x32 f16: lane m=L&15, hi=L>>4; two contiguous 8-half runs -> 2x b128.
// B 32x16 f16: lane = column; staged in fragment order -> 2x ds_load_b128.
// C/D 16x16 f32: VGPR r, lane L -> m = r + 8*(L>>4), n = L&15.
// ---------------------------------------------------------------------------
__device__ __forceinline__ v16h load_a_frag(const _Float16* __restrict__ wk, int hi) {
  const v8h a0 = *(const v8h*)(wk + hi * 16);
  const v8h a1 = *(const v8h*)(wk + hi * 16 + 8);
  v16h a;
#pragma unroll
  for (int i = 0; i < 8; ++i) { a[i] = a0[i]; a[8 + i] = a1[i]; }
  return a;
}

__device__ __forceinline__ v16h load_b_frag(const _Float16* __restrict__ bp) {
  const v8h b0 = *(const v8h*)(bp);
  const v8h b1 = *(const v8h*)(bp + 8);
  v16h b;
#pragma unroll
  for (int i = 0; i < 8; ++i) { b[i] = b0[i]; b[8 + i] = b1[i]; }
  return b;
}

// ---------------------------------------------------------------------------
// Implicit-GEMM conv, tap-major K (k = tap*IC + ic), f16 WMMA, f32 accum.
// N-tile 64 (4 B-fragments / 4 accumulators per wave, one shared A load).
// Software pipeline: double-buffered LDS, gather(next)->wmma(cur)->scatter(next),
// one barrier per K-step; next-tile global loads overlap current WMMAs.
// MODE 0: 3x3 conv pad=1 (stride templated).  MODE 1: 4x4 s2 p1 deconv.
// ---------------------------------------------------------------------------
template <int IC, int OC, int OCV, int IH, int IW, int STRIDE, int MODE,
          bool RELU, bool F32OUT>
__global__ __launch_bounds__((OC / 16) * 32)
void conv_wmma(const _Float16* __restrict__ in, const _Float16* __restrict__ wgt,
               const float* __restrict__ bias, void* __restrict__ out) {
  constexpr int OH = (MODE == 0) ? ((IH - 1) / STRIDE + 1) : IH * 2;
  constexpr int OW = (MODE == 0) ? ((IW - 1) / STRIDE + 1) : IW * 2;
  constexpr int NP = OH * OW;
  constexpr int TAPS = (MODE == 0) ? 9 : 16;
  constexpr int K  = IC * TAPS;            // IC % 32 == 0 for all layers
  constexpr int NT = 64;
  constexpr int TOT = 32 * NT;             // staged halves per tile
  constexpr int BD  = (OC / 16) * 32;      // blockDim.x (compile-time)
  constexpr int ELEMS = (TOT + BD - 1) / BD;
  constexpr bool EXACT = (TOT % BD == 0);
  constexpr int LOG2OW = (OW == 64) ? 6 : 5;
  static_assert(OW == 32 || OW == 64, "OW must be power of two");

  __shared__ __attribute__((aligned(32))) _Float16 Bs[2][TOT];  // ping-pong

  const int img  = blockIdx.y;
  const int nt   = blockIdx.x * NT;
  const int wave = (int)(threadIdx.x >> 5);
  const int lane = (int)(threadIdx.x & 31);
  const int hi   = lane >> 4;
  const int mrow = lane & 15;

  const _Float16* __restrict__ inimg = in + (size_t)img * IC * IH * IW;
  const _Float16* __restrict__ wrow  = wgt + (size_t)(wave * 16 + mrow) * K;

  _Float16 vreg[ELEMS];

  // Gather one im2col tile (tap_, icb_) into registers.
  auto gather = [&](int tap_, int icb_) {
    const int fy = (MODE == 0) ? (tap_ / 3) : (tap_ >> 2);
    const int fx = (MODE == 0) ? (tap_ % 3) : (tap_ & 3);
#pragma unroll
    for (int e = 0; e < ELEMS; ++e) {
      const int t = (int)threadIdx.x + e * BD;
      _Float16 v = (_Float16)0.0f;
      if (EXACT || t < TOT) {
        const int kk = t >> 6;             // 0..31 (ic offset)
        const int nn = t & 63;             // 0..63 (coalesced in x)
        const int p  = nt + nn;
        const int oy = p >> LOG2OW;
        const int ox = p & (OW - 1);
        if (MODE == 0) {
          const int iy = oy * STRIDE - 1 + fy;
          const int ix = ox * STRIDE - 1 + fx;
          if (iy >= 0 && iy < IH && ix >= 0 && ix < IW)
            v = inimg[(size_t)(icb_ + kk) * (IH * IW) + iy * IW + ix];
        } else {
          const int iy2 = oy + fy - 2;
          const int ix2 = ox + fx - 2;
          if (iy2 >= 0 && ix2 >= 0 && ((iy2 & 1) == 0) && ((ix2 & 1) == 0)) {
            const int iy = iy2 >> 1, ix = ix2 >> 1;
            if (iy < IH && ix < IW)
              v = inimg[(size_t)(icb_ + kk) * (IH * IW) + iy * IW + ix];
          }
        }
      }
      vreg[e] = v;
    }
  };

  // Scatter registers into fragment-order LDS tile.
  auto scatter = [&](_Float16* dst) {
#pragma unroll
    for (int e = 0; e < ELEMS; ++e) {
      const int t = (int)threadIdx.x + e * BD;
      if (EXACT || t < TOT) {
        const int kk = t >> 6;
        const int nn = t & 63;
        const int s = nn >> 4, col = nn & 15;
        dst[((s * 32 + (kk >> 4) * 16 + col) << 4) | (kk & 15)] = vreg[e];
      }
    }
  };

  v8f acc[4] = {};

  // Prologue: stage tile 0.
  gather(0, 0);
  scatter(Bs[0]);
  __syncthreads();

  int cur = 0, tap = 0, icb = 0;
  for (int k0 = 0; k0 < K; k0 += 32) {
    int ntap = tap, nicb = icb + 32;
    if (nicb == IC) { nicb = 0; ++ntap; }
    const bool more = (k0 + 32 < K);

    if (more) gather(ntap, nicb);           // next tile: global loads in flight

    const v16h a = load_a_frag(wrow + k0, hi);
#pragma unroll
    for (int s = 0; s < 4; ++s) {
      const v16h b = load_b_frag(&Bs[cur][(s * 32 + lane) << 4]);
      acc[s] = __builtin_amdgcn_wmma_f32_16x16x32_f16(false, a, false, b,
                                                      (short)0, acc[s], false, false);
    }

    if (more) scatter(&Bs[cur ^ 1][0]);     // commit next tile after WMMAs
    __syncthreads();                        // single barrier per K-step
    cur ^= 1; tap = ntap; icb = nicb;
  }

#pragma unroll
  for (int s = 0; s < 4; ++s) {
#pragma unroll
    for (int r = 0; r < 8; ++r) {
      const int oc = wave * 16 + r + 8 * hi;
      if (OCV == OC || oc < OCV) {          // folds away when OCV == OC
        float v = acc[s][r] + bias[oc];
        if (RELU) v = fmaxf(v, 0.0f);
        const size_t o = ((size_t)img * OCV + oc) * NP +
                         (size_t)(nt + s * 16 + (lane & 15));
        if (F32OUT) ((float*)out)[o]    = v;
        else        ((_Float16*)out)[o] = (_Float16)v;
      }
    }
  }
}

// ---------------------------------------------------------------------------
// DAP (49x49 projection, LDS-cached) + softmax expectation + coords add.
// cost layout [b*49 + d][HW] == (b, D, h, w), d = v*7 + u.
// 48 blocks x 128 threads == 2*HW pixels exactly.
// ---------------------------------------------------------------------------
__global__ void dap_flow_kernel(const float* __restrict__ cost,
                                const float* __restrict__ wdap,
                                const float* __restrict__ coords,
                                float* __restrict__ out) {
  __shared__ float wsm[49 * 49];
  for (int t = threadIdx.x; t < 49 * 49; t += blockDim.x) wsm[t] = wdap[t];
  __syncthreads();

  const int idx = blockIdx.x * blockDim.x + threadIdx.x;  // 0 .. 6143
  const int b = idx / HW_;
  const int p = idx % HW_;

  float cin[49];
  for (int d = 0; d < 49; ++d) cin[d] = cost[((size_t)b * 49 + d) * HW_ + p];

  float cpr[49];
  float mx = -3.4e38f;
  for (int e = 0; e < 49; ++e) {
    float s = 0.f;
    for (int d = 0; d < 49; ++d) s += wsm[e * 49 + d] * cin[d];
    cpr[e] = s;
    mx = fmaxf(mx, s);
  }
  float sum = 0.f;
  for (int e = 0; e < 49; ++e) { const float t = expf(cpr[e] - mx); cpr[e] = t; sum += t; }
  const float inv = 1.0f / sum;
  float dx = 0.f, dy = 0.f;
  for (int e = 0; e < 49; ++e) {
    const float pr = cpr[e] * inv;
    dx += pr * (float)((e % 7) - 3);
    dy += pr * (float)((e / 7) - 3);
  }
  out[((size_t)b * 2 + 0) * HW_ + p] = coords[((size_t)b * 2 + 0) * HW_ + p] + dx;
  out[((size_t)b * 2 + 1) * HW_ + p] = coords[((size_t)b * 2 + 1) * HW_ + p] + dy;
}

// ---------------------------------------------------------------------------
// Launcher
// ---------------------------------------------------------------------------
extern "C" void kernel_launch(void* const* d_in, const int* in_sizes, int n_in,
                              void* d_out, int out_size, void* d_ws, size_t ws_size,
                              hipStream_t stream) {
  (void)in_sizes; (void)n_in; (void)out_size; (void)ws_size;

  const float* feat1 = (const float*)d_in[0];
  const float* feat2 = (const float*)d_in[1];
  const float* coords = (const float*)d_in[2];
  const float* w0 = (const float*)d_in[3];  const float* b0 = (const float*)d_in[4];
  const float* w1 = (const float*)d_in[5];  const float* b1 = (const float*)d_in[6];
  const float* w2 = (const float*)d_in[7];  const float* b2 = (const float*)d_in[8];
  const float* w3 = (const float*)d_in[9];  const float* b3 = (const float*)d_in[10];
  const float* wd = (const float*)d_in[11]; const float* bd = (const float*)d_in[12];
  const float* w5 = (const float*)d_in[13]; const float* b5 = (const float*)d_in[14];
  const float* wdap = (const float*)d_in[15];

  char* ws = (char*)d_ws;
  size_t off = 0;
  auto alloc = [&](size_t nbytes) -> void* {
    void* p = ws + off;
    off += (nbytes + 255) & ~(size_t)255;
    return p;
  };

  // f16 weight buffers (tap-major K order; deconv/conv5 padded to 64 rows)
  _Float16* W0b = (_Float16*)alloc((size_t)96 * 1152 * 2);
  _Float16* W1b = (_Float16*)alloc((size_t)128 * 864 * 2);
  _Float16* W2b = (_Float16*)alloc((size_t)128 * 1152 * 2);
  _Float16* W3b = (_Float16*)alloc((size_t)64 * 1152 * 2);
  _Float16* WDb = (_Float16*)alloc((size_t)64 * 1024 * 2);
  _Float16* W5b = (_Float16*)alloc((size_t)64 * 288 * 2);

  // Activation regions (ping-pong): A: feat / x2, B: x0 / x3, C: x1 / xd
  _Float16* regA = (_Float16*)alloc((size_t)98 * 128 * 3072 * 2);  // 77 MB
  _Float16* regB = (_Float16*)alloc((size_t)98 * 96 * 3072 * 2);   // 58 MB
  _Float16* regC = (_Float16*)alloc((size_t)98 * 128 * 768 * 2);   // 19 MB
  float*    cost = (float*)alloc((size_t)98 * 3072 * 4);           // 1.2 MB

  auto perm = [&](const float* s, _Float16* d, int IC, int TAPS, int OCV, int ndst) {
    permute_w_f16<<<(ndst + 255) / 256, 256, 0, stream>>>(s, d, IC, TAPS, OCV, ndst);
  };
  perm(w0, W0b, 128, 9, 96, 96 * 1152);
  perm(w1, W1b, 96, 9, 128, 128 * 864);
  perm(w2, W2b, 128, 9, 128, 128 * 1152);
  perm(w3, W3b, 128, 9, 64, 64 * 1152);
  perm(wd, WDb, 64, 16, 32, 64 * 1024);   // pad OC 32 -> 64
  perm(w5, W5b, 32, 9, 1, 64 * 288);      // pad OC 1  -> 64

  // Cost-volume input: feat = concat(feat1, warp(feat2)) in f16
  warp_concat_kernel<<<(98 * HW_) / 256, 256, 0, stream>>>(feat1, feat2, coords, regA);

  // MatchingNet (N-tile 64 => grid.x = NP/64)
  conv_wmma<128, 96, 96, 48, 64, 1, 0, true, false>
      <<<dim3(48, 98), 6 * 32, 0, stream>>>(regA, W0b, b0, regB);
  conv_wmma<96, 128, 128, 48, 64, 2, 0, true, false>
      <<<dim3(12, 98), 8 * 32, 0, stream>>>(regB, W1b, b1, regC);
  conv_wmma<128, 128, 128, 24, 32, 1, 0, true, false>
      <<<dim3(12, 98), 8 * 32, 0, stream>>>(regC, W2b, b2, regA);
  conv_wmma<128, 64, 64, 24, 32, 1, 0, true, false>
      <<<dim3(12, 98), 4 * 32, 0, stream>>>(regA, W3b, b3, regB);
  conv_wmma<64, 64, 32, 24, 32, 1, 1, true, false>      // deconv 4x4 s2 p1
      <<<dim3(48, 98), 4 * 32, 0, stream>>>(regB, WDb, bd, regC);
  conv_wmma<32, 64, 1, 48, 64, 1, 0, false, true>
      <<<dim3(48, 98), 4 * 32, 0, stream>>>(regC, W5b, b5, cost);

  // DAP + softmax flow regression -> d_out = coords + delta
  dap_flow_kernel<<<48, 128, 0, stream>>>(cost, wdap, coords, (float*)d_out);
}
